// TransformerBlock_89163521065124
// MI455X (gfx1250) — compile-verified, matrix operand order checked by reading, not run on hardware
//
#include <hip/hip_runtime.h>

typedef __attribute__((ext_vector_type(16))) __bf16 v16bf;
typedef __attribute__((ext_vector_type(8)))  float  v8f;

constexpr int kN  = 50000;   // nodes
constexpr int kE  = 800000;  // edges
constexpr int kD  = 128;     // in channels
constexpr int kH  = 8;       // heads
constexpr int kC  = 16;      // head dim
constexpr int kED = 32;      // edge dim
constexpr int kHC = 128;     // H*C
constexpr int kFH = 512;     // 4*D (ffn hidden) == qkvs width

// ---------------------------------------------------------------------------
// Weight preparation: transpose + f32->bf16 so WMMA B fragments load
// contiguously along K.  wcat = [Wq|Wk|Wv|Wskip]^T as [512 out][128 k],
// wet = We^T [128][32], w1t = W1^T [512][128], w2t = W2^T [128][512].
// ---------------------------------------------------------------------------
__global__ void prep_weights(
    const float* __restrict__ Wq, const float* __restrict__ Wk,
    const float* __restrict__ Wv, const float* __restrict__ Wskip,
    const float* __restrict__ bq, const float* __restrict__ bk,
    const float* __restrict__ bv, const float* __restrict__ bskip,
    const float* __restrict__ We, const float* __restrict__ W1,
    const float* __restrict__ W2,
    __bf16* __restrict__ wcat, float* __restrict__ bcat,
    __bf16* __restrict__ wet,  __bf16* __restrict__ w1t,
    __bf16* __restrict__ w2t)
{
  int i = blockIdx.x * blockDim.x + threadIdx.x;
  if (i < 512 * 128) {                       // wcat
    int n = i >> 7, k = i & 127;
    int sel = n >> 7, c = n & 127;
    const float* W = (sel == 0) ? Wq : (sel == 1) ? Wk : (sel == 2) ? Wv : Wskip;
    wcat[i] = (__bf16)W[k * kHC + c];
    if (k == 0) {
      const float* bb = (sel == 0) ? bq : (sel == 1) ? bk : (sel == 2) ? bv : bskip;
      bcat[n] = bb[c];
    }
  } else if (i < 512 * 128 + 128 * 32) {     // wet
    int j = i - 512 * 128;
    int n = j >> 5, k = j & 31;
    wet[j] = (__bf16)We[k * kHC + n];
  } else if (i < 512 * 128 + 128 * 32 + 512 * 128) {   // w1t
    int j = i - (512 * 128 + 128 * 32);
    int n = j >> 7, k = j & 127;
    w1t[j] = (__bf16)W1[k * kFH + n];
  } else if (i < 512 * 128 + 128 * 32 + 512 * 128 + 128 * 512) {  // w2t
    int j = i - (512 * 128 + 128 * 32 + 512 * 128);
    int n = j >> 9, k = j & 511;
    w2t[j] = (__bf16)W2[k * kD + n];
  }
}

__global__ void init_buffers(float* __restrict__ mmax, float* __restrict__ denom,
                             float* __restrict__ attn)
{
  int i = blockIdx.x * blockDim.x + threadIdx.x;
  if (i < kN * kH) { mmax[i] = -__builtin_inff(); denom[i] = 0.f; }
  if (i < kN * kHC) attn[i] = 0.f;
}

// ---------------------------------------------------------------------------
// Generic wave32 WMMA GEMM:  C[M x (16*NT)] = A[M x K] * Bt^T + bias
// A is f32 row-major (converted to bf16 on the fly), Bt is bf16 [n][k].
// One wave owns one 16-row M tile and loops over 16-col N tiles.
// Fragment layout per CDNA5 ISA 7.12.2 (16-bit A/B, 16x16x32):
//   lanes 0-15  -> row/col = lane,     K offsets {kb+0..7,  kb+16..23}
//   lanes 16-31 -> row/col = lane-16,  K offsets {kb+8..15, kb+24..31}
// f32 C/D: vector elem r -> row = base + (lane<16?0:8) + r, col = lane&15.
// For K<=128 the A fragments are loaded+converted ONCE per wave (<=32 VGPRs)
// and reused across all N tiles; for K=512 they are re-streamed from L1 to
// keep register pressure / occupancy sane.
// ---------------------------------------------------------------------------
__device__ __forceinline__ v16bf load_a_frag(const float* __restrict__ pa0)
{
  v16bf a;
#pragma unroll
  for (int j = 0; j < 8; ++j) {
    a[j]     = (__bf16)pa0[j];
    a[8 + j] = (__bf16)pa0[16 + j];
  }
  return a;
}

__device__ __forceinline__ v16bf load_b_frag(const __bf16* __restrict__ pb0)
{
  v16bf b;
#pragma unroll
  for (int j = 0; j < 8; ++j) {
    b[j]     = pb0[j];
    b[8 + j] = pb0[16 + j];
  }
  return b;
}

template<int K, int NT, bool RELU, bool OUTBF16>
__global__ void wmma_gemm(const float* __restrict__ A, int lda,
                          const __bf16* __restrict__ Bt,
                          const float* __restrict__ bias,
                          void* __restrict__ Cout, int ldc,
                          int Mtiles)
{
  constexpr int KT = K / 32;
  int wv    = threadIdx.x >> 5;
  int mtile = blockIdx.x * (blockDim.x >> 5) + wv;
  if (mtile >= Mtiles) return;               // whole-wave uniform exit
  int lane = threadIdx.x & 31;
  int hf   = lane >> 4;
  int lr   = lane & 15;
  const float* arow = A + (size_t)(mtile * 16 + lr) * lda;

  v16bf afrag[(K <= 128) ? KT : 1];
  if constexpr (K <= 128) {
#pragma unroll
    for (int kb = 0; kb < KT; ++kb)
      afrag[kb] = load_a_frag(arow + kb * 32 + hf * 8);
  }

  for (int nt = 0; nt < NT; ++nt) {
    int col = nt * 16 + lr;
    const __bf16* brow = Bt + (size_t)col * K;
    float bv = bias ? bias[col] : 0.f;
    v8f acc;
#pragma unroll
    for (int r = 0; r < 8; ++r) acc[r] = bv;

#pragma unroll
    for (int kb = 0; kb < KT; ++kb) {
      v16bf a;
      if constexpr (K <= 128) a = afrag[kb];
      else                    a = load_a_frag(arow + kb * 32 + hf * 8);
      v16bf b = load_b_frag(brow + kb * 32 + hf * 8);
      acc = __builtin_amdgcn_wmma_f32_16x16x32_bf16(
          false, a, false, b, (short)0, acc, false, false);
    }
    if constexpr (RELU) {
#pragma unroll
      for (int r = 0; r < 8; ++r) acc[r] = fmaxf(acc[r], 0.f);
    }
    size_t rowbase = (size_t)(mtile * 16 + hf * 8);
    if constexpr (OUTBF16) {
      __bf16* cp = (__bf16*)Cout + rowbase * ldc + col;
#pragma unroll
      for (int r = 0; r < 8; ++r) cp[(size_t)r * ldc] = (__bf16)acc[r];
    } else {
      float* cp = (float*)Cout + rowbase * ldc + col;
#pragma unroll
      for (int r = 0; r < 8; ++r) cp[(size_t)r * ldc] = acc[r];
    }
  }
}

// ---------------------------------------------------------------------------
// Edge attention (segment softmax via atomics; q/k/v tables are L2 resident)
// ---------------------------------------------------------------------------
__device__ inline void atomicMaxF(float* addr, float v) {
  if (v >= 0.f) atomicMax((int*)addr, __float_as_int(v));
  else          atomicMin((unsigned int*)addr, (unsigned int)__float_as_int(v));
}

__global__ void edge_alpha(const int* __restrict__ eidx,
                           const float* __restrict__ qkvs,
                           const __bf16* __restrict__ eproj,
                           float* __restrict__ alpha,
                           float* __restrict__ mmax)
{
  int e = blockIdx.x * blockDim.x + threadIdx.x;
  if (e >= kE) return;
  size_t s = (size_t)eidx[e];
  size_t d = (size_t)eidx[kE + e];
  const float*  q  = qkvs + d * kFH;         // q cols 0..127
  const float*  k  = qkvs + s * kFH + 128;   // k cols 128..255
  const __bf16* ev = eproj + (size_t)e * kHC;
#pragma unroll
  for (int h = 0; h < kH; ++h) {
    float acc = 0.f;
#pragma unroll
    for (int c = 0; c < kC; ++c) {
      int idx = h * kC + c;
      acc += q[idx] * (k[idx] + (float)ev[idx]);
    }
    acc *= 0.25f;                            // 1/sqrt(C), C=16
    alpha[(size_t)e * kH + h] = acc;
    atomicMaxF(&mmax[d * kH + h], acc);
  }
}

__global__ void edge_p(const int* __restrict__ eidx,
                       float* __restrict__ alpha,
                       const float* __restrict__ mmax,
                       float* __restrict__ denom)
{
  int e = blockIdx.x * blockDim.x + threadIdx.x;
  if (e >= kE) return;
  size_t d = (size_t)eidx[kE + e];
#pragma unroll
  for (int h = 0; h < kH; ++h) {
    float p = __expf(alpha[(size_t)e * kH + h] - mmax[d * kH + h]);
    alpha[(size_t)e * kH + h] = p;
    unsafeAtomicAdd(&denom[d * kH + h], p);
  }
}

__global__ void edge_agg(const int* __restrict__ eidx,
                         const float* __restrict__ qkvs,
                         const __bf16* __restrict__ eproj,
                         const float* __restrict__ p,
                         const float* __restrict__ denom,
                         float* __restrict__ attn)
{
  int e = blockIdx.x * blockDim.x + threadIdx.x;
  if (e >= kE) return;
  size_t s = (size_t)eidx[e];
  size_t d = (size_t)eidx[kE + e];
  const float*  v  = qkvs + s * kFH + 256;   // v cols 256..383
  const __bf16* ev = eproj + (size_t)e * kHC;
  float* out = attn + d * kHC;
#pragma unroll
  for (int h = 0; h < kH; ++h) {
    float a = p[(size_t)e * kH + h] / (denom[d * kH + h] + 1e-16f);
#pragma unroll
    for (int c = 0; c < kC; ++c) {
      int idx = h * kC + c;
      unsafeAtomicAdd(&out[idx], (v[idx] + (float)ev[idx]) * a);
    }
  }
}

// ---------------------------------------------------------------------------
// LayerNorm over 128 features, one wave per row; optional strided third
// addend (skip slice of qkvs) + its bias.  out = LN(a + b [+ c + cbias]).
// Row bases are 512B aligned -> use float4 (b128) accesses.
// ---------------------------------------------------------------------------
__global__ void ln_kernel(const float* __restrict__ a, const float* __restrict__ b,
                          const float* __restrict__ c, int cld, int coff,
                          const float* __restrict__ cbias,
                          const float* __restrict__ g, const float* __restrict__ beta,
                          float* __restrict__ out, int n_rows)
{
  int wv = threadIdx.x >> 5;
  int n  = blockIdx.x * (blockDim.x >> 5) + wv;
  if (n >= n_rows) return;
  int lane = threadIdx.x & 31;
  int ci   = lane * 4;
  size_t base = (size_t)n * kD + ci;
  float4 va = *(const float4*)(a + base);
  float4 vb = *(const float4*)(b + base);
  float p0 = va.x + vb.x, p1 = va.y + vb.y, p2 = va.z + vb.z, p3 = va.w + vb.w;
  if (c) {
    float4 vc = *(const float4*)(c + (size_t)n * cld + coff + ci);
    float4 vs4 = *(const float4*)(cbias + ci);
    p0 += vc.x + vs4.x; p1 += vc.y + vs4.y; p2 += vc.z + vs4.z; p3 += vc.w + vs4.w;
  }
  float s = p0 + p1 + p2 + p3;
#pragma unroll
  for (int off = 16; off > 0; off >>= 1) s += __shfl_xor(s, off, 32);
  float mu = s * (1.0f / kD);
  float d0 = p0 - mu, d1 = p1 - mu, d2 = p2 - mu, d3 = p3 - mu;
  float vs = d0 * d0 + d1 * d1 + d2 * d2 + d3 * d3;
#pragma unroll
  for (int off = 16; off > 0; off >>= 1) vs += __shfl_xor(vs, off, 32);
  float rs = rsqrtf(vs * (1.0f / kD) + 1e-5f);
  float4 vg = *(const float4*)(g + ci);
  float4 vbt = *(const float4*)(beta + ci);
  float4 o;
  o.x = d0 * rs * vg.x + vbt.x;
  o.y = d1 * rs * vg.y + vbt.y;
  o.z = d2 * rs * vg.z + vbt.z;
  o.w = d3 * rs * vg.w + vbt.w;
  *(float4*)(out + base) = o;
}

// ---------------------------------------------------------------------------
extern "C" void kernel_launch(void* const* d_in, const int* in_sizes, int n_in,
                              void* d_out, int out_size, void* d_ws, size_t ws_size,
                              hipStream_t stream)
{
  (void)in_sizes; (void)n_in; (void)out_size; (void)ws_size;
  const float* x     = (const float*)d_in[0];
  const int*   eidx  = (const int*)  d_in[1];
  const float* eattr = (const float*)d_in[2];
  const float* Wq    = (const float*)d_in[3];
  const float* bq    = (const float*)d_in[4];
  const float* Wk    = (const float*)d_in[5];
  const float* bk    = (const float*)d_in[6];
  const float* Wv    = (const float*)d_in[7];
  const float* bv    = (const float*)d_in[8];
  const float* We    = (const float*)d_in[9];
  const float* Wsk   = (const float*)d_in[10];
  const float* bsk   = (const float*)d_in[11];
  const float* ln1g  = (const float*)d_in[12];
  const float* ln1b  = (const float*)d_in[13];
  const float* W1    = (const float*)d_in[14];
  const float* b1    = (const float*)d_in[15];
  const float* W2    = (const float*)d_in[16];
  const float* b2    = (const float*)d_in[17];
  const float* ln2g  = (const float*)d_in[18];
  const float* ln2b  = (const float*)d_in[19];

  char* wp = (char*)d_ws;
  auto take = [&](size_t bytes) -> char* {
    char* r = wp; wp += (bytes + 255) & ~(size_t)255; return r;
  };
  __bf16* wcat  = (__bf16*)take(512 * 128 * sizeof(__bf16));
  float*  bcat  = (float*) take(512 * sizeof(float));
  __bf16* wet   = (__bf16*)take(128 * 32 * sizeof(__bf16));
  __bf16* w1t   = (__bf16*)take(512 * 128 * sizeof(__bf16));
  __bf16* w2t   = (__bf16*)take(128 * 512 * sizeof(__bf16));
  float*  qkvs  = (float*) take((size_t)kN * kFH * sizeof(float));   // later: ffn hidden
  __bf16* eproj = (__bf16*)take((size_t)kE * kHC * sizeof(__bf16));
  float*  alpha = (float*) take((size_t)kE * kH * sizeof(float));
  float*  mmax  = (float*) take((size_t)kN * kH * sizeof(float));
  float*  denom = (float*) take((size_t)kN * kH * sizeof(float));
  float*  attn  = (float*) take((size_t)kN * kHC * sizeof(float));   // later: ffn out
  float*  h     = (float*) take((size_t)kN * kD * sizeof(float));

  // 1) weight prep + accumulator init
  prep_weights<<<784, 256, 0, stream>>>(Wq, Wk, Wv, Wsk, bq, bk, bv, bsk,
                                        We, W1, W2, wcat, bcat, wet, w1t, w2t);
  init_buffers<<<25000, 256, 0, stream>>>(mmax, denom, attn);

  // 2) fused q|k|v|skip projection:  [50000,128] x [128,512]
  wmma_gemm<128, 32, false, false><<<391, 256, 0, stream>>>(
      x, kD, wcat, bcat, qkvs, kFH, kN / 16);
  // 3) edge projection (bf16 out):   [800000,32] x [32,128]
  wmma_gemm<32, 8, false, true><<<6250, 256, 0, stream>>>(
      eattr, kED, wet, nullptr, eproj, kHC, kE / 16);
  // 4) attention: alpha + segment max / exp + segment sum / weighted scatter
  edge_alpha<<<3125, 256, 0, stream>>>(eidx, qkvs, eproj, alpha, mmax);
  edge_p<<<3125, 256, 0, stream>>>(eidx, alpha, mmax, denom);
  edge_agg<<<3125, 256, 0, stream>>>(eidx, qkvs, eproj, alpha, denom, attn);

  // 5) h = LN1(x + attn + skip + bskip)
  ln_kernel<<<6250, 256, 0, stream>>>(x, attn, qkvs, kFH, 384, bsk,
                                      ln1g, ln1b, h, kN);
  // 6) FFN1 (ReLU):  [50000,128] x [128,512] -> hid (reuses qkvs)
  wmma_gemm<128, 32, true, false><<<391, 256, 0, stream>>>(
      h, kD, w1t, b1, qkvs, kFH, kN / 16);
  // 7) FFN2:         [50000,512] x [512,128] -> ffn out (reuses attn)
  wmma_gemm<512, 8, false, false><<<391, 256, 0, stream>>>(
      qkvs, kFH, w2t, b2, attn, kHC, kN / 16);
  // 8) out = LN2(h + ffn)
  ln_kernel<<<6250, 256, 0, stream>>>(h, attn, nullptr, 0, 0, nullptr,
                                      ln2g, ln2b, (float*)d_out, kN);
}
